// ExtractionBlock_41704132444909
// MI455X (gfx1250) — matrix operand order repaired and checked by reading, not compiled, not run on hardware
//
#include <hip/hip_runtime.h>
#include <hip/hip_bf16.h>

// ---------------------------------------------------------------------------
// Problem constants
// ---------------------------------------------------------------------------
#define BATCH   8
#define NPTS    8192
#define C_IN    64
#define C_OUT   128
#define NSAMP   2048
#define KNN_K   32

typedef __attribute__((ext_vector_type(16))) __bf16 v16bf;
typedef __attribute__((ext_vector_type(8)))  float  v8f;

// LDS row stride (in bf16 elements) for the [n][c] activation tiles.
#define XT_STRIDE 132

// ---------------------------------------------------------------------------
// Workspace layout (bytes)
// ---------------------------------------------------------------------------
#define WBF_OFF   0                         // 9 * 128*128 bf16       = 294912
#define SB_OFF    294912                    // 9 * 256 f32            = 9216
#define FPS_OFF   304128                    // 8 * 2048 int           = 65536
#define KNN_OFF   369664                    // 8 * 2048 * 32 int      = 2097152
#define POOL_OFF  2466816                   // 8 * 2048 * 128 f32     = 8388608
// total ~10.9 MB

struct PtrPack { const float* p[27]; };

// ---------------------------------------------------------------------------
// Kernel 0: fold BN into scale/bias, convert the 9 conv weights to bf16
// conv order: te, pre_w1[0], pre_w2[0], pre_w1[1], pre_w2[1],
//             pos_w1[0], pos_w2[0], pos_w1[1], pos_w2[1]
// ---------------------------------------------------------------------------
__global__ __launch_bounds__(256) void prep_kernel(PtrPack P, __bf16* wbf, float* sb) {
    const int t = threadIdx.x;
    const float* W[9] = { P.p[2], P.p[7],        P.p[8],        P.p[7]+16384,  P.p[8]+16384,
                          P.p[17],       P.p[18],       P.p[17]+16384, P.p[18]+16384 };
    const float* G[9] = { P.p[3], P.p[9],  P.p[10], P.p[9]+128,  P.p[10]+128,
                          P.p[19], P.p[20], P.p[19]+128, P.p[20]+128 };
    const float* Bb[9]= { P.p[4], P.p[11], P.p[12], P.p[11]+128, P.p[12]+128,
                          P.p[21], P.p[22], P.p[21]+128, P.p[22]+128 };
    const float* M[9] = { P.p[5], P.p[13], P.p[14], P.p[13]+128, P.p[14]+128,
                          P.p[23], P.p[24], P.p[23]+128, P.p[24]+128 };
    const float* V[9] = { P.p[6], P.p[15], P.p[16], P.p[15]+128, P.p[16]+128,
                          P.p[25], P.p[26], P.p[25]+128, P.p[26]+128 };
    for (int cv = 0; cv < 9; ++cv) {
        const float* w = W[cv];
        __bf16* d = wbf + cv * 16384;
        for (int i = t; i < 16384; i += 256) d[i] = (__bf16)w[i];
        if (t < 128) {
            float sc = G[cv][t] * __frsqrt_rn(V[cv][t] + 1e-5f);
            sb[cv * 256 + t]       = sc;
            sb[cv * 256 + 128 + t] = Bb[cv][t] - M[cv][t] * sc;
        }
    }
}

// ---------------------------------------------------------------------------
// Kernel 1: farthest point sampling (one workgroup per batch, serial steps)
// Writes fps indices and new_xyz (first output).
// ---------------------------------------------------------------------------
__global__ __launch_bounds__(1024) void fps_kernel(const float* __restrict__ xyz,
                                                   int* __restrict__ fps,
                                                   float* __restrict__ out_xyz) {
    const int b = blockIdx.x;
    const int t = threadIdx.x;
    float px[8], py[8], pz[8], dist[8];
#pragma unroll
    for (int i = 0; i < 8; ++i) {
        const int p = t + i * 1024;
        const float* r = xyz + ((size_t)b * NPTS + p) * 3;
        px[i] = r[0]; py[i] = r[1]; pz[i] = r[2];
        dist[i] = 1e10f;
    }
    __shared__ float lp[3];
    __shared__ int   cur;
    __shared__ float rv[1024];
    __shared__ int   ri[1024];
    if (t == 0) cur = 0;
    __syncthreads();
    for (int s = 0; s < NSAMP; ++s) {
        if (t == 0) {
            const int c = cur;
            const float* r = xyz + ((size_t)b * NPTS + c) * 3;
            lp[0] = r[0]; lp[1] = r[1]; lp[2] = r[2];
            fps[b * NSAMP + s] = c;
            float* o = out_xyz + ((size_t)b * NSAMP + s) * 3;
            o[0] = r[0]; o[1] = r[1]; o[2] = r[2];
        }
        __syncthreads();
        const float bx = lp[0], by = lp[1], bz = lp[2];
        float bestv = -1.0f; int besti = 0x7fffffff;
#pragma unroll
        for (int i = 0; i < 8; ++i) {
            const float dx = px[i] - bx, dy = py[i] - by, dz = pz[i] - bz;
            const float d = dx * dx + dy * dy + dz * dz;
            dist[i] = fminf(dist[i], d);
            const int p = t + i * 1024;
            if (dist[i] > bestv || (dist[i] == bestv && p < besti)) { bestv = dist[i]; besti = p; }
        }
        rv[t] = bestv; ri[t] = besti;
        __syncthreads();
        for (int off = 512; off > 0; off >>= 1) {
            if (t < off) {
                const float v2 = rv[t + off]; const int i2 = ri[t + off];
                if (v2 > rv[t] || (v2 == rv[t] && i2 < ri[t])) { rv[t] = v2; ri[t] = i2; }
            }
            __syncthreads();
        }
        if (t == 0) cur = ri[0];
        __syncthreads();
    }
}

// ---------------------------------------------------------------------------
// Kernel 2: KNN (one workgroup per query; d^2 in LDS, 32 argmin passes)
// ---------------------------------------------------------------------------
__global__ __launch_bounds__(256) void knn_kernel(const float* __restrict__ xyz,
                                                  const int* __restrict__ fps,
                                                  int* __restrict__ knn) {
    __shared__ float d2[NPTS];
    __shared__ float rv[256];
    __shared__ int   ri[256];
    const int g = blockIdx.x;
    const int b = g >> 11;
    const int s = g & (NSAMP - 1);
    const int t = threadIdx.x;
    const int qi = fps[b * NSAMP + s];
    const float* q = xyz + ((size_t)b * NPTS + qi) * 3;
    const float qx = q[0], qy = q[1], qz = q[2];
    const float* base = xyz + (size_t)b * NPTS * 3;
#pragma unroll 4
    for (int i = 0; i < 32; ++i) {
        const int p = t + i * 256;
        const float dx = base[p * 3 + 0] - qx;
        const float dy = base[p * 3 + 1] - qy;
        const float dz = base[p * 3 + 2] - qz;
        d2[p] = dx * dx + dy * dy + dz * dz;
    }
    __syncthreads();
    for (int kk = 0; kk < KNN_K; ++kk) {
        float bestv = 1e38f; int besti = 0x7fffffff;
#pragma unroll 4
        for (int i = 0; i < 32; ++i) {
            const int p = t + i * 256;
            const float v = d2[p];
            if (v < bestv || (v == bestv && p < besti)) { bestv = v; besti = p; }
        }
        rv[t] = bestv; ri[t] = besti;
        __syncthreads();
        for (int off = 128; off > 0; off >>= 1) {
            if (t < off) {
                const float v2 = rv[t + off]; const int i2 = ri[t + off];
                if (v2 < rv[t] || (v2 == rv[t] && i2 < ri[t])) { rv[t] = v2; ri[t] = i2; }
            }
            __syncthreads();
        }
        if (t == 0) { knn[(size_t)g * KNN_K + kk] = ri[0]; d2[ri[0]] = 1e38f; }
        __syncthreads();
    }
}

// ---------------------------------------------------------------------------
// WMMA conv layer: out[o][n] = relu( scale[o]*sum_c W[o][c]*Xin[n][c] + bias[o]
//                                    (+ skip[n][o]) )
// Xin/Xout/Xskip are LDS tiles in [n][c] order, row stride XT_STRIDE (bf16).
// 8 waves; wave w owns output rows 16w..16w+15; NT = number of 16-col N-tiles.
// Fragment packing follows the CDNA5 16x16x32 bf16 VGPR layouts.
// ---------------------------------------------------------------------------
template <int NT>
__device__ inline void conv128(const __bf16* __restrict__ W,      // 128x128 row-major, global
                               const __bf16* __restrict__ Xin,    // LDS
                               __bf16* __restrict__ Xout,         // LDS
                               const __bf16* __restrict__ Xskip,  // LDS or nullptr
                               const float* __restrict__ scale,
                               const float* __restrict__ bias) {
    const int lane = threadIdx.x & 31;
    const int wv   = threadIdx.x >> 5;   // 0..7: output row tile
    const int m    = lane & 15;
    const int hl   = lane >> 4;          // lane half selects K sub-block
#pragma unroll
    for (int nt = 0; nt < NT; ++nt) {
        v8f acc = {};
        const int n = nt * 16 + m;       // B/D column owned by this lane
#pragma unroll
        for (int kk = 0; kk < 4; ++kk) { // K = 128 in 4 steps of 32
            v16bf av, bv;
            const __bf16* wrow = W + (size_t)(wv * 16 + m) * 128 + kk * 32;
#pragma unroll
            for (int v = 0; v < 4; ++v) { // A: k = hl*8 + 2v (v<4), 16+hl*8+2(v-4)
                const int k = hl * 8 + 2 * v;
                av[2 * v]     = wrow[k];
                av[2 * v + 1] = wrow[k + 1];
            }
#pragma unroll
            for (int v = 4; v < 8; ++v) {
                const int k = 16 + hl * 8 + 2 * (v - 4);
                av[2 * v]     = wrow[k];
                av[2 * v + 1] = wrow[k + 1];
            }
            const __bf16* xrow = Xin + n * XT_STRIDE + kk * 32 + hl * 8;
#pragma unroll
            for (int v = 0; v < 8; ++v) { // B: k = hl*8 + 2v (v<4), 16+hl*8+2(v-4)
                bv[2 * v]     = xrow[2 * v + (v >= 4 ? 8 : 0)];
                bv[2 * v + 1] = xrow[2 * v + 1 + (v >= 4 ? 8 : 0)];
            }
            acc = __builtin_amdgcn_wmma_f32_16x16x32_bf16(
                false, av, false, bv, (short)0, acc, false, false);
        }
        // epilogue: BN fold + optional residual + relu, store transposed [n][o]
#pragma unroll
        for (int v = 0; v < 8; ++v) {    // D: row = v + 8*hl within tile
            const int o = wv * 16 + hl * 8 + v;
            float val = acc[v] * scale[o] + bias[o];
            if (Xskip) val += (float)Xskip[n * XT_STRIDE + o];
            val = val > 0.0f ? val : 0.0f;
            Xout[n * XT_STRIDE + o] = (__bf16)val;
        }
    }
}

// ---------------------------------------------------------------------------
// Kernel 3: per-group gather + te conv + 2 residual blocks + max-pool over K
// One workgroup (256 thr / 8 waves) per (b, s) group. Activations 128x32.
// ---------------------------------------------------------------------------
__global__ __launch_bounds__(256) void group_mlp_kernel(const float* __restrict__ feat,
                                                        const int* __restrict__ fps,
                                                        const int* __restrict__ knn,
                                                        const __bf16* __restrict__ wbf,
                                                        const float* __restrict__ sb,
                                                        float* __restrict__ pooled) {
    __shared__ __bf16 X[3][KNN_K * XT_STRIDE];
    __shared__ int kidx[KNN_K];
    __shared__ int ctr;
    const int g = blockIdx.x;
    const int b = g >> 11;
    const int s = g & (NSAMP - 1);
    const int t = threadIdx.x;
    if (t < KNN_K) kidx[t] = knn[(size_t)g * KNN_K + t];
    if (t == 0)    ctr = fps[b * NSAMP + s];
    __syncthreads();
    { // stage X[0][k][c]: c<64 -> grouped neighbor feat, c>=64 -> center feat
        const int k  = t >> 3;
        const int c0 = (t & 7) * 16;
        const float* frow = feat + ((size_t)b * NPTS + kidx[k]) * C_IN;
        const float* crow = feat + ((size_t)b * NPTS + ctr) * C_IN;
        __bf16* dst = &X[0][k * XT_STRIDE];
#pragma unroll
        for (int j = 0; j < 16; ++j) {
            const int c = c0 + j;
            const float v = (c < C_IN) ? frow[c] : crow[c - C_IN];
            dst[c] = (__bf16)v;
        }
    }
    __syncthreads();
    conv128<2>(wbf + 0 * 16384, X[0], X[1], nullptr, sb + 0 * 256, sb + 0 * 256 + 128);
    __syncthreads();
    conv128<2>(wbf + 1 * 16384, X[1], X[2], nullptr, sb + 1 * 256, sb + 1 * 256 + 128);
    __syncthreads();
    conv128<2>(wbf + 2 * 16384, X[2], X[0], X[1],    sb + 2 * 256, sb + 2 * 256 + 128);
    __syncthreads();
    conv128<2>(wbf + 3 * 16384, X[0], X[1], nullptr, sb + 3 * 256, sb + 3 * 256 + 128);
    __syncthreads();
    conv128<2>(wbf + 4 * 16384, X[1], X[2], X[0],    sb + 4 * 256, sb + 4 * 256 + 128);
    __syncthreads();
    if (t < C_OUT) { // max over the K axis
        float mx = 0.0f; // post-relu values are >= 0
        for (int k = 0; k < KNN_K; ++k) mx = fmaxf(mx, (float)X[2][k * XT_STRIDE + t]);
        pooled[(size_t)g * C_OUT + t] = mx;
    }
}

// ---------------------------------------------------------------------------
// Kernel 4: pos residual blocks over (B, 128, 2048); 64-sample tiles.
// Writes x (second output) in (B, C, S) order.
// ---------------------------------------------------------------------------
__global__ __launch_bounds__(256) void pos_mlp_kernel(const float* __restrict__ pooled,
                                                      const __bf16* __restrict__ wbf,
                                                      const float* __restrict__ sb,
                                                      float* __restrict__ outx) {
    __shared__ __bf16 X[3][64 * XT_STRIDE];
    const int b  = blockIdx.x >> 5;
    const int s0 = (blockIdx.x & 31) * 64;
    const int t  = threadIdx.x;
    { // stage X[0][sl][c] from pooled[b][s0+sl][c]
        const int sl = t >> 2;
        const int c0 = (t & 3) * 32;
        const float* src = pooled + ((size_t)b * NSAMP + s0 + sl) * C_OUT + c0;
        __bf16* dst = &X[0][sl * XT_STRIDE + c0];
#pragma unroll
        for (int j = 0; j < 32; ++j) dst[j] = (__bf16)src[j];
    }
    __syncthreads();
    conv128<4>(wbf + 5 * 16384, X[0], X[1], nullptr, sb + 5 * 256, sb + 5 * 256 + 128);
    __syncthreads();
    conv128<4>(wbf + 6 * 16384, X[1], X[2], X[0],    sb + 6 * 256, sb + 6 * 256 + 128);
    __syncthreads();
    conv128<4>(wbf + 7 * 16384, X[2], X[0], nullptr, sb + 7 * 256, sb + 7 * 256 + 128);
    __syncthreads();
    conv128<4>(wbf + 8 * 16384, X[0], X[1], X[2],    sb + 8 * 256, sb + 8 * 256 + 128);
    __syncthreads();
    // write out[b][c][s0+sl]
    for (int j = 0; j < 32; ++j) {
        const int idx = t * 32 + j;
        const int sl  = idx & 63;
        const int c   = idx >> 6;
        outx[((size_t)b * C_OUT + c) * NSAMP + s0 + sl] = (float)X[1][sl * XT_STRIDE + c];
    }
}

// ---------------------------------------------------------------------------
// Host launcher
// ---------------------------------------------------------------------------
extern "C" void kernel_launch(void* const* d_in, const int* in_sizes, int n_in,
                              void* d_out, int out_size, void* d_ws, size_t ws_size,
                              hipStream_t stream) {
    (void)in_sizes; (void)n_in; (void)out_size; (void)ws_size;
    const float* xyz  = (const float*)d_in[0];
    const float* feat = (const float*)d_in[1];

    char* ws = (char*)d_ws;
    __bf16* wbf    = (__bf16*)(ws + WBF_OFF);
    float*  sb     = (float*)(ws + SB_OFF);
    int*    fpsIdx = (int*)(ws + FPS_OFF);
    int*    knnIdx = (int*)(ws + KNN_OFF);
    float*  pooled = (float*)(ws + POOL_OFF);

    float* out_xyz = (float*)d_out;                       // (8, 2048, 3)
    float* out_x   = (float*)d_out + BATCH * NSAMP * 3;   // (8, 128, 2048)

    PtrPack P;
    for (int i = 0; i < 27; ++i) P.p[i] = (const float*)d_in[i];

    hipLaunchKernelGGL(prep_kernel, dim3(1), dim3(256), 0, stream, P, wbf, sb);
    hipLaunchKernelGGL(fps_kernel, dim3(BATCH), dim3(1024), 0, stream, xyz, fpsIdx, out_xyz);
    hipLaunchKernelGGL(knn_kernel, dim3(BATCH * NSAMP), dim3(256), 0, stream, xyz, fpsIdx, knnIdx);
    hipLaunchKernelGGL(group_mlp_kernel, dim3(BATCH * NSAMP), dim3(256), 0, stream,
                       feat, fpsIdx, knnIdx, wbf, sb, pooled);
    hipLaunchKernelGGL(pos_mlp_kernel, dim3(BATCH * 32), dim3(256), 0, stream,
                       pooled, wbf, sb, out_x);
}
